// BinaryLayer_59528246722736
// MI455X (gfx1250) — 8-bit matrix operand order repaired and checked by reading, compiled, not run on hardware
//
#include <hip/hip_runtime.h>
#include <stdint.h>

typedef int      v16i __attribute__((ext_vector_type(16)));
typedef float    v8f  __attribute__((ext_vector_type(8)));
typedef uint32_t v4u  __attribute__((ext_vector_type(4)));
typedef uint32_t v8u  __attribute__((ext_vector_type(8)));

static constexpr int M  = 8192;
static constexpr int K  = 4096;
static constexpr int N  = 4096;

static constexpr int BM = 256;
static constexpr int BN = 128;
static constexpr int BK = 128;
static constexpr int LDT = BK + 16;

__global__ __launch_bounds__(256) void k_binarize_rows(const float* __restrict__ in,
                                                       unsigned char* __restrict__ out,
                                                       int n4)
{
    int i = blockIdx.x * 256 + threadIdx.x;
    if (i >= n4) return;
    float4 v = ((const float4*)in)[i];
    uint32_t b =  ((v.x >= 0.0f) ? 0x38u : 0xB8u)
               | (((v.y >= 0.0f) ? 0x38u : 0xB8u) << 8)
               | (((v.z >= 0.0f) ? 0x38u : 0xB8u) << 16)
               | (((v.w >= 0.0f) ? 0x38u : 0xB8u) << 24);
    ((uint32_t*)out)[i] = b;
}

__global__ __launch_bounds__(256) void k_binarize_transpose(const float* __restrict__ in,
                                                            unsigned char* __restrict__ out)
{
    __shared__ unsigned char tile[32][33];
    const int tx = threadIdx.x & 31;
    const int ty = threadIdx.x >> 5;
    const int k0 = blockIdx.y * 32;
    const int n0 = blockIdx.x * 32;

    #pragma unroll
    for (int r = 0; r < 4; ++r) {
        int k = ty + r * 8;
        float v = in[(size_t)(k0 + k) * N + (n0 + tx)];
        tile[k][tx] = (v >= 0.0f) ? 0x38 : 0xB8;
    }
    __syncthreads();

    const int n  = threadIdx.x >> 3;
    const int kk = (threadIdx.x & 7) * 4;
    uint32_t b =  (uint32_t)tile[kk + 0][n]
               | ((uint32_t)tile[kk + 1][n] << 8)
               | ((uint32_t)tile[kk + 2][n] << 16)
               | ((uint32_t)tile[kk + 3][n] << 24);
    *(uint32_t*)&out[(size_t)(n0 + n) * K + (k0 + kk)] = b;
}

__device__ __forceinline__ void tdm_load_2d(uint32_t lds_addr, const unsigned char* gptr,
                                            uint32_t tile_d0_bytes, uint32_t tile_d1,
                                            uint32_t tensor_d0, uint32_t tensor_d1,
                                            uint32_t stride_bytes)
{
    uint64_t ga = (uint64_t)(uintptr_t)gptr;
    v4u g0;
    g0.x = 1u;
    g0.y = lds_addr;
    g0.z = (uint32_t)ga;
    g0.w = ((uint32_t)(ga >> 32) & 0x01FFFFFFu) | (2u << 30);
    v8u g1;
    g1[0] = (1u << 20) | (4u << 22) | (3u << 25);
    g1[1] = (tensor_d0 & 0xFFFFu) << 16;
    g1[2] = ((tensor_d0 >> 16) & 0xFFFFu) | ((tensor_d1 & 0xFFFFu) << 16);
    g1[3] = ((tensor_d1 >> 16) & 0xFFFFu) | ((tile_d0_bytes & 0xFFFFu) << 16);
    g1[4] = (tile_d1 & 0xFFFFu);
    g1[5] = stride_bytes;
    g1[6] = 0u;
    g1[7] = 0u;
    asm volatile("tensor_load_to_lds %0, %1" :: "s"(g0), "s"(g1) : "memory");
}

__global__ __launch_bounds__(256) void k_bgemm_fp8(const unsigned char* __restrict__ A8,
                                                   const unsigned char* __restrict__ Bt,
                                                   float* __restrict__ out)
{
    __shared__ __align__(16) unsigned char As[2][BM * LDT];
    __shared__ __align__(16) unsigned char Bs[2][BN * LDT];

    const int tid   = threadIdx.x;
    const int lane  = tid & 31;
    const int wave  = tid >> 5;
    const int waveM = wave >> 1;
    const int waveN = wave & 1;

    const int blockM = blockIdx.y * BM;
    const int blockN = blockIdx.x * BN;

    const uint32_t ldsA[2] = { (uint32_t)(uintptr_t)&As[0][0], (uint32_t)(uintptr_t)&As[1][0] };
    const uint32_t ldsB[2] = { (uint32_t)(uintptr_t)&Bs[0][0], (uint32_t)(uintptr_t)&Bs[1][0] };

    const unsigned char* gA = A8 + (size_t)blockM * K;
    const unsigned char* gB = Bt + (size_t)blockN * K;

    const int r16 = lane & 15;
    const int hiA = (lane >> 4) * 16;
    const int hiB = (lane >> 4) * 16;

    v8f acc[4][4];
    #pragma unroll
    for (int i = 0; i < 4; ++i)
        #pragma unroll
        for (int j = 0; j < 4; ++j)
            #pragma unroll
            for (int e = 0; e < 8; ++e) acc[i][j][e] = 0.0f;

    if (tid < 32) {
        tdm_load_2d(ldsA[0], gA, BK, BM, K, M, K);
        tdm_load_2d(ldsB[0], gB, BK, BN, K, N, K);
    }

    int buf = 0;
    for (int k0 = 0; k0 < K; k0 += BK) {
        if (tid < 32) {
            if (k0 + BK < K) {
                tdm_load_2d(ldsA[buf ^ 1], gA + k0 + BK, BK, BM, K, M, K);
                tdm_load_2d(ldsB[buf ^ 1], gB + k0 + BK, BK, BN, K, N, K);
                __builtin_amdgcn_s_wait_tensorcnt(2);
            } else {
                __builtin_amdgcn_s_wait_tensorcnt(0);
            }
        }
        __syncthreads();

        const unsigned char* Ab = As[buf];
        const unsigned char* Bb = Bs[buf];

        v16i bfrag[4];
        #pragma unroll
        for (int tj = 0; tj < 4; ++tj) {
            const unsigned char* base = Bb + (waveN * 64 + tj * 16 + r16) * LDT;
            #pragma unroll
            for (int j = 0; j < 4; ++j) {
                uint4 q = *(const uint4*)(base + 32 * j + hiB);
                bfrag[tj][4 * j + 0] = (int)q.x;
                bfrag[tj][4 * j + 1] = (int)q.y;
                bfrag[tj][4 * j + 2] = (int)q.z;
                bfrag[tj][4 * j + 3] = (int)q.w;
            }
        }
        #pragma unroll
        for (int ti = 0; ti < 4; ++ti) {
            v16i afrag;
            const unsigned char* base = Ab + (waveM * 64 + ti * 16 + r16) * LDT;
            #pragma unroll
            for (int j = 0; j < 4; ++j) {
                uint4 q = *(const uint4*)(base + 32 * j + hiA);
                afrag[4 * j + 0] = (int)q.x;
                afrag[4 * j + 1] = (int)q.y;
                afrag[4 * j + 2] = (int)q.z;
                afrag[4 * j + 3] = (int)q.w;
            }
            #pragma unroll
            for (int tj = 0; tj < 4; ++tj)
                acc[ti][tj] = __builtin_amdgcn_wmma_f32_16x16x128_fp8_fp8(
                    afrag, bfrag[tj], (short)0, acc[ti][tj], false, false);
        }

        __syncthreads();
        buf ^= 1;
    }

    const int ccol  = lane & 15;
    const int rbase = (lane >> 4) * 8;
    #pragma unroll
    for (int ti = 0; ti < 4; ++ti)
        #pragma unroll
        for (int tj = 0; tj < 4; ++tj) {
            const int gcol = blockN + waveN * 64 + tj * 16 + ccol;
            #pragma unroll
            for (int v = 0; v < 8; ++v) {
                const int grow = blockM + waveM * 64 + ti * 16 + rbase + v;
                out[(size_t)grow * N + gcol] = acc[ti][tj][v];
            }
        }
}

extern "C" void kernel_launch(void* const* d_in, const int* in_sizes, int n_in,
                              void* d_out, int out_size, void* d_ws, size_t ws_size,
                              hipStream_t stream)
{
    (void)in_sizes; (void)n_in; (void)out_size; (void)ws_size;
    const float* x = (const float*)d_in[0];
    const float* w = (const float*)d_in[1];
    float* out = (float*)d_out;

    unsigned char* A8 = (unsigned char*)d_ws;
    unsigned char* Bt = A8 + (size_t)M * K;

    const int n4x = M * K / 4;
    k_binarize_rows<<<(n4x + 255) / 256, 256, 0, stream>>>(x, A8, n4x);
    k_binarize_transpose<<<dim3(N / 32, K / 32), 256, 0, stream>>>(w, Bt);
    k_bgemm_fp8<<<dim3(N / BN, M / BM), 256, 0, stream>>>(A8, Bt, out);
}
